// QuantizedLinearCUDA_19550691131798
// MI455X (gfx1250) — compile-verified, matrix operand order checked by reading, not run on hardware
//
#include <hip/hip_runtime.h>
#include <hip/hip_bf16.h>

typedef _Float16 v16h __attribute__((ext_vector_type(16)));
typedef _Float16 v8h  __attribute__((ext_vector_type(8)));
typedef _Float16 h2   __attribute__((ext_vector_type(2)));
typedef float    v8f  __attribute__((ext_vector_type(8)));
typedef float    v4f  __attribute__((ext_vector_type(4)));
typedef int      v4i  __attribute__((ext_vector_type(4)));

// ---------------------------------------------------------------------------
// int4 pair -> packed f16 pair, sign-corrected, via mantissa-magic:
//   t = b ^ 0x88 flips nibble sign bits; (nib | 0x6400) as f16 = 1024 + nib;
//   subtract 1032 -> (nib^8) - 8 = signed int4.  One v_pk_add_f16 per byte.
// ---------------------------------------------------------------------------
__device__ __forceinline__ unsigned dec_byte(int b) {
    unsigned t = (unsigned)b ^ 0x88u;
    union { unsigned u; h2 h; } cv;
    cv.u = (t & 0xFu) | ((t & 0xF0u) << 12) | 0x64006400u;
    const h2 off = { (_Float16)(-1032.0f), (_Float16)(-1032.0f) };
    cv.h = cv.h + off;
    return cv.u;
}

// ---------------------------------------------------------------------------
// Kernel 1: per-row correction  c[m] = sum_g s[g]*zp[g] * sum_{k in g} x[m,k]
// ---------------------------------------------------------------------------
__global__ void int4lin_rowcorr(const float* __restrict__ x,
                                const float* __restrict__ scales,
                                const float* __restrict__ zps,
                                float* __restrict__ c,
                                int K, int G) {
    const int m = blockIdx.x;
    const float* xr = x + (size_t)m * K;
    float acc = 0.f;
    for (int k = threadIdx.x; k < K; k += blockDim.x) {
        const int g = k / G;
        acc += xr[k] * scales[g] * zps[g];
    }
    __shared__ float red[256];
    red[threadIdx.x] = acc;
    __syncthreads();
    for (int s = 128; s > 0; s >>= 1) {
        if ((int)threadIdx.x < s) red[threadIdx.x] += red[threadIdx.x + s];
        __syncthreads();
    }
    if (threadIdx.x == 0) c[m] = red[0];
}

// ---------------------------------------------------------------------------
// Kernel 2a: x16[m,k] = f16( x[m,k] * s[k/G] )   (8 elems / thread)
// ---------------------------------------------------------------------------
__global__ __launch_bounds__(256)
void int4lin_cvt_x16(const float* __restrict__ x,
                     const float* __restrict__ scales,
                     _Float16* __restrict__ x16, int K, int G) {
    const size_t i8 = ((size_t)blockIdx.x * 256 + threadIdx.x) * 8;
    const float s = scales[(int)(i8 % (size_t)K) / G];   // 8 | G: one group
    v4f a = *(const v4f*)(x + i8);
    v4f b = *(const v4f*)(x + i8 + 4);
    v8h h;
    #pragma unroll
    for (int i = 0; i < 4; ++i) h[i]     = (_Float16)(a[i] * s);
    #pragma unroll
    for (int i = 0; i < 4; ++i) h[4 + i] = (_Float16)(b[i] * s);
    *(v8h*)(x16 + i8) = h;
}

// ---------------------------------------------------------------------------
// Kernel 2b: w16[n,k] = f16(signed int4)  from packed bytes (8 bytes/thread)
// ---------------------------------------------------------------------------
__global__ __launch_bounds__(256)
void int4lin_cvt_w16(const int* __restrict__ pw,
                     _Float16* __restrict__ w16) {
    const size_t j8 = ((size_t)blockIdx.x * 256 + threadIdx.x) * 8;
    v4i a = *(const v4i*)(pw + j8);
    v4i b = *(const v4i*)(pw + j8 + 4);
    union { unsigned u[4]; v8h h; } p0, p1;
    #pragma unroll
    for (int i = 0; i < 4; ++i) p0.u[i] = dec_byte(a[i]);
    #pragma unroll
    for (int i = 0; i < 4; ++i) p1.u[i] = dec_byte(b[i]);
    *(v8h*)(w16 + 2 * j8)     = p0.h;
    *(v8h*)(w16 + 2 * j8 + 8) = p1.h;
}

// ---------------------------------------------------------------------------
// Kernel 3: WMMA GEMM  out = A @ B^T - c[m] + bias[n]
//   PRE=true : A = x16 [M,K] f16, B = w16 [N,K] f16 (pure f16 GEMM, L2-resident)
//   PRE=false: fused path (f32 x * scale, packed int4 decode) in the loop
//   BM=BN=128, BK=64, 256 threads = 8 waves; each wave 64x32 (4x2 tiles)
//   Double-buffered LDS, one barrier per K-slab, 16 WMMAs per barrier.
// ---------------------------------------------------------------------------
#define BM 128
#define BN 128
#define BK 64
#define LDSTRIDE 72   // halves; 144 bytes/row, multiple of 16B

template <bool PRE>
__global__ __launch_bounds__(256)
void int4lin_wmma_gemm(const float* __restrict__ x,        // [M,K] f32 (PRE=false)
                       const int*   __restrict__ pw,       // [N,K/2]   (PRE=false)
                       const _Float16* __restrict__ x16,   // [M,K] f16 (PRE=true)
                       const _Float16* __restrict__ w16,   // [N,K] f16 (PRE=true)
                       const float* __restrict__ scales,   // [K/G]     (PRE=false)
                       const float* __restrict__ bias,     // [N]
                       const float* __restrict__ crow,     // [M]
                       float* __restrict__ out,            // [M,N]
                       int M, int N, int K, int G) {
    __shared__ __align__(16) _Float16 As[2][BM][LDSTRIDE];
    __shared__ __align__(16) _Float16 Ws[2][BN][LDSTRIDE];

    const int t      = threadIdx.x;
    const int lane   = t & 31;
    const int wave   = t >> 5;
    const int wave_m = wave & 1;
    const int wave_n = wave >> 1;
    const int m0 = blockIdx.y * BM;
    const int n0 = blockIdx.x * BN;
    const int Khalf = K >> 1;

    const int row  = t >> 1;            // loader row (both A and W tiles)
    const int colh = (t & 1) * 32;      // 32 halves per thread
    const int colf = (t & 1) * 32;      // 32 floats per thread (fused A)
    const int colp = (t & 1) * 16;      // 16 packed int32 per thread (fused W)

    // fragment addressing (ISA 7.12.2): lanes 0-15 K-base 0, lanes 16-31 K-base 8
    const int frow = lane & 15;
    const int kb   = (lane >> 4) * 8;

    v8f acc[4][2] = {};
    const int nk = K / BK;

    // staging registers
    v8h ah[4], wh[4];                   // PRE path
    v4f xr[8]; v4i wr[4]; float s_tile; // fused path

    const _Float16* ag0 = x16 + (size_t)(m0 + row) * K + colh;
    const _Float16* bg0 = w16 + (size_t)(n0 + row) * K + colh;
    const float*    xg0 = x   + (size_t)(m0 + row) * K + colf;
    const int*      wg0 = pw  + (size_t)(n0 + row) * Khalf + colp;

    auto load_slab = [&](int kt) {
        if constexpr (PRE) {
            const _Float16* ag = ag0 + kt * BK;
            const _Float16* bg = bg0 + kt * BK;
            #pragma unroll
            for (int i = 0; i < 4; ++i) ah[i] = *(const v8h*)(ag + 8 * i);
            #pragma unroll
            for (int i = 0; i < 4; ++i) wh[i] = *(const v8h*)(bg + 8 * i);
        } else {
            const float* xg = xg0 + kt * BK;
            const int*   wg = wg0 + kt * (BK / 2);
            #pragma unroll
            for (int i = 0; i < 8; ++i) xr[i] = *(const v4f*)(xg + 4 * i);
            #pragma unroll
            for (int i = 0; i < 4; ++i) wr[i] = *(const v4i*)(wg + 4 * i);
            s_tile = scales[(kt * BK) / G];
        }
    };

    auto store_slab = [&](int p) {
        if constexpr (PRE) {
            #pragma unroll
            for (int j = 0; j < 4; ++j) *(v8h*)&As[p][row][colh + 8 * j] = ah[j];
            #pragma unroll
            for (int j = 0; j < 4; ++j) *(v8h*)&Ws[p][row][colh + 8 * j] = wh[j];
        } else {
            #pragma unroll
            for (int j = 0; j < 4; ++j) {
                v8h h;
                #pragma unroll
                for (int i = 0; i < 4; ++i) h[i]     = (_Float16)(xr[2*j][i]   * s_tile);
                #pragma unroll
                for (int i = 0; i < 4; ++i) h[4 + i] = (_Float16)(xr[2*j+1][i] * s_tile);
                *(v8h*)&As[p][row][colf + 8 * j] = h;
            }
            #pragma unroll
            for (int j = 0; j < 4; ++j) {
                union { unsigned u[4]; v8h h; } pk;
                #pragma unroll
                for (int i = 0; i < 4; ++i) pk.u[i] = dec_byte(wr[j][i]);
                *(v8h*)&Ws[p][row][2 * colp + 8 * j] = pk.h;
            }
        }
    };

    auto compute_slab = [&](int p) {
        #pragma unroll
        for (int s = 0; s < 2; ++s) {
            const int ko = s * 32;
            v16h afr[4], bfr[2];
            #pragma unroll
            for (int fm = 0; fm < 4; ++fm) {
                const int mr = wave_m * 64 + fm * 16 + frow;
                ((v8h*)&afr[fm])[0] = *(const v8h*)&As[p][mr][ko + kb];
                ((v8h*)&afr[fm])[1] = *(const v8h*)&As[p][mr][ko + kb + 16];
            }
            #pragma unroll
            for (int fn = 0; fn < 2; ++fn) {
                const int nr = wave_n * 32 + fn * 16 + frow;
                ((v8h*)&bfr[fn])[0] = *(const v8h*)&Ws[p][nr][ko + kb];
                ((v8h*)&bfr[fn])[1] = *(const v8h*)&Ws[p][nr][ko + kb + 16];
            }
            #pragma unroll
            for (int fm = 0; fm < 4; ++fm)
                #pragma unroll
                for (int fn = 0; fn < 2; ++fn)
                    acc[fm][fn] = __builtin_amdgcn_wmma_f32_16x16x32_f16(
                        false, afr[fm], false, bfr[fn],
                        (short)0, acc[fm][fn], false, false);
        }
    };

    load_slab(0);
    store_slab(0);
    __syncthreads();

    for (int kt = 0; kt < nk; ++kt) {
        const int p = kt & 1;
        if (kt + 1 < nk) {
            load_slab(kt + 1);
            if (kt + 2 < nk) {
                if constexpr (PRE) {
                    __builtin_prefetch(ag0 + (kt + 2) * BK, 0, 3);
                    __builtin_prefetch(bg0 + (kt + 2) * BK, 0, 3);
                } else {
                    __builtin_prefetch(xg0 + (kt + 2) * BK, 0, 3);
                    __builtin_prefetch(wg0 + (kt + 2) * (BK / 2), 0, 3);
                }
            }
        }
        compute_slab(p);
        if (kt + 1 < nk) {
            store_slab(1 - p);
            __syncthreads();
        }
    }

    // epilogue: acc - c[m] + bias[n]  (C/D: VGPR r -> rows r / r+8)
    const int halfsel = lane >> 4;
    const int nl      = lane & 15;
    #pragma unroll
    for (int fm = 0; fm < 4; ++fm) {
        const int mbase = m0 + wave_m * 64 + fm * 16 + halfsel * 8;
        float cv[8];
        #pragma unroll
        for (int r = 0; r < 8; ++r) cv[r] = crow[mbase + r];
        #pragma unroll
        for (int fn = 0; fn < 2; ++fn) {
            const int nn = n0 + wave_n * 32 + fn * 16 + nl;
            const float bv = bias[nn];
            #pragma unroll
            for (int r = 0; r < 8; ++r)
                out[(size_t)(mbase + r) * N + nn] = acc[fm][fn][r] - cv[r] + bv;
        }
    }
}

// ---------------------------------------------------------------------------
extern "C" void kernel_launch(void* const* d_in, const int* in_sizes, int n_in,
                              void* d_out, int out_size, void* d_ws, size_t ws_size,
                              hipStream_t stream) {
    const float* x      = (const float*)d_in[0];
    const int*   pw     = (const int*)d_in[1];
    const float* scales = (const float*)d_in[2];
    const float* zps    = (const float*)d_in[3];
    const float* bias   = (const float*)d_in[4];
    float* out = (float*)d_out;

    const int N = in_sizes[4];                       // bias
    const int K = (int)(2LL * in_sizes[1] / N);      // packed_w = N * K/2
    const int M = in_sizes[0] / K;                   // x = M * K
    const int G = K / in_sizes[2];                   // scales = K/G

    // workspace layout: c [M f32] | x16 [M*K f16] | w16 [N*K f16]
    float* c = (float*)d_ws;
    const size_t off_x16 = (size_t)M * 4;
    const size_t off_w16 = off_x16 + (size_t)M * K * 2;
    const size_t need    = off_w16 + (size_t)N * K * 2;
    _Float16* x16 = (_Float16*)((char*)d_ws + off_x16);
    _Float16* w16 = (_Float16*)((char*)d_ws + off_w16);

    int4lin_rowcorr<<<M, 256, 0, stream>>>(x, scales, zps, c, K, G);

    dim3 grid(N / BN, M / BM);
    if (ws_size >= need) {
        const size_t ex = (size_t)M * K / 8;         // 8 elems per thread
        const size_t ew = (size_t)N * (K / 2) / 8;   // 8 bytes per thread
        int4lin_cvt_x16<<<(unsigned)(ex / 256), 256, 0, stream>>>(x, scales, x16, K, G);
        int4lin_cvt_w16<<<(unsigned)(ew / 256), 256, 0, stream>>>(pw, w16);
        int4lin_wmma_gemm<true><<<grid, 256, 0, stream>>>(
            x, pw, x16, w16, scales, bias, c, out, M, N, K, G);
    } else {
        int4lin_wmma_gemm<false><<<grid, 256, 0, stream>>>(
            x, pw, x16, w16, scales, bias, c, out, M, N, K, G);
    }
}
